// FeedbackTransformerModel_60181081751899
// MI455X (gfx1250) — matrix operand order repaired and checked by reading, not compiled, not run on hardware
//
#include <hip/hip_runtime.h>

// ---------------------------------------------------------------------------
// Feedback Transformer persistent kernel for MI455X (gfx1250, wave32, WMMA)
// B=8, S=512, V=1024, D=1024, H=16, L=4, F=4096, HD=64
// All WMMA operands (A and B) are kept pre-swizzled in fragment layout so the
// inner K loop is: 2x b128 A-loads + 2x b128 W-loads + 1x v_wmma. No branches,
// no conversions in the loop.
// ---------------------------------------------------------------------------

#define NBLK 64
#define NTHR 256

typedef __attribute__((ext_vector_type(16))) __bf16        v16bf;
typedef __attribute__((ext_vector_type(8)))  float         v8f;
typedef __attribute__((ext_vector_type(4)))  unsigned int  v4u;

constexpr int Bb = 8, Ss = 512, Vv = 1024, Dd = 1024, Ll = 4, Ff = 4096, HDd = 64;
constexpr float SCALE = 0.125f;          // 1/sqrt(64)
constexpr float INV_D = 1.0f / 1024.0f;

struct Args {
  const int* ids;
  const float *emb, *pe;
  const float *q_w, *q_b, *out_w, *out_b;
  const float *ff1_w, *ff1_b, *ff2_w, *ff2_b;
  const float *n1_s, *n1_b, *n2_s, *n2_b;
  const float *mem_w, *k_w, *k_b, *v_w, *v_b, *lm_w, *lm_b;
  float* out;
  // workspace partitions
  unsigned* bar;
  __bf16 *wq, *wout, *wff1, *wff2, *wk, *wv, *wlm;   // swizzled bf16 weights (B-frag)
  float *xseq, *mk, *mv;                             // embedded input, recurrent memory
  float *qbuf, *y1, *st1, *y2, *st2;
  __bf16 *attb, *a1b;                                // A-fragment-swizzled activations
  int steps, layers;
};

union Bfr { v16bf v; v4u u[2]; };

// ---------------------------------------------------------------------------
// A-fragment layout: 16x32 bf16 tile kt stored as 512 contiguous bf16;
// lane = (kh<<4)|m owns 16 contiguous bf16 (32B). Inverse map (m,k) -> offset:
// ---------------------------------------------------------------------------
__device__ inline int afrag_off(int m, int k) {
  const int k32 = k & 31;
  const int kh = (k32 >> 3) & 1;
  const int e = (k32 & 7) + ((k32 & 16) ? 8 : 0);
  return (k >> 5) * 512 + (kh * 16 + m) * 16 + e;
}

// ---------------------------------------------------------------------------
// grid-wide split barrier (counter + generation), spin with s_sleep
// ---------------------------------------------------------------------------
__device__ inline void gsync(unsigned* bar) {
  __syncthreads();
  if (threadIdx.x == 0) {
    __threadfence();
    volatile unsigned* gen = bar + 64;
    unsigned g = *gen;
    unsigned arrived = atomicAdd(bar, 1u);
    if (arrived == (unsigned)(NBLK - 1)) {
      bar[0] = 0u;
      __threadfence();
      atomicAdd((unsigned*)gen, 1u);
    } else {
      while (*gen == g) __builtin_amdgcn_s_sleep(1);
    }
    __threadfence();
  }
  __syncthreads();
}

// ---------------------------------------------------------------------------
// One 16-wide output tile of  C(8xN) = A(8xK) @ W(KxN) : K split over 8 waves,
// cross-wave reduce in LDS, epilogue on 128 threads. Optional LN-stats atomics.
// Afrag: A in fragment layout (LDS or global, generic pointer).
// W: pre-swizzled B-fragment tiles; tile (nt,kt) is 512 contiguous bf16.
// ---------------------------------------------------------------------------
template <class Epi>
__device__ inline void run_tile(const __bf16* __restrict__ Afrag,
                                const __bf16* __restrict__ W, int KT, int nt,
                                float* sred, float* stats, Epi epi) {
  const int tid = threadIdx.x;
  const int wave = tid >> 5, lane = tid & 31;
  v8f c = {};
  const int ktper = KT >> 3;
  const int kt0 = wave * ktper;
  const v4u* wl = (const v4u*)(W + ((size_t)nt * KT + kt0) * 512) + lane * 2;
  const v4u* al = (const v4u*)(Afrag + (size_t)kt0 * 512) + lane * 2;
#pragma unroll 2
  for (int i = 0; i < ktper; ++i) {
    Bfr b, a;
    b.u[0] = wl[0];
    b.u[1] = wl[1];
    a.u[0] = al[0];
    a.u[1] = al[1];
    __builtin_prefetch((const void*)(wl + 64), 0, 1);   // next 1KB W tile
    c = __builtin_amdgcn_wmma_f32_16x16x32_bf16(false, a.v, false, b.v,
                                                (short)0, c, false, false);
    wl += 64;
    al += 64;
  }
  if (lane < 16) {
#pragma unroll
    for (int r = 0; r < 8; ++r) sred[(wave * 8 + r) * 16 + lane] = c[r];
  }
  __syncthreads();
  if (tid < 128) {
    const int r = tid >> 4, n = tid & 15;
    float acc = 0.f;
#pragma unroll
    for (int w = 0; w < 8; ++w) acc += sred[(w * 8 + r) * 16 + n];
    float y = epi(r, nt * 16 + n, acc);
    if (stats) {
      sred[r * 16 + n]       = y;       // reuse wave0 region for y
      sred[(8 + r) * 16 + n] = y * y;   // wave1 region for y^2
    }
  }
  __syncthreads();
  if (stats && tid < 8) {
    float s1 = 0.f, s2 = 0.f;
#pragma unroll
    for (int n = 0; n < 16; ++n) {
      s1 += sred[tid * 16 + n];
      s2 += sred[(8 + tid) * 16 + n];
    }
    atomicAdd(stats + tid * 2 + 0, s1);
    atomicAdd(stats + tid * 2 + 1, s2);
  }
  __syncthreads();
}

// ---------------------------------------------------------------------------
// LayerNorm of one element, from (y, sum, sumsq) stats
// ---------------------------------------------------------------------------
__device__ inline float ln_at(const float* y, const float* st, const float* sc,
                              const float* bi, int m, int d) {
  float mu  = st[m * 2] * INV_D;
  float var = st[m * 2 + 1] * INV_D - mu * mu;
  return (y[m * 1024 + d] - mu) * rsqrtf(var + 1e-5f) * sc[d] + bi[d];
}

__device__ inline float hcur_at(int l, int t, const Args& A, int m, int d) {
  if (l == 0) return A.xseq[(size_t)t * 8192 + m * 1024 + d];
  return ln_at(A.y2 + (size_t)(l - 1) * 8192, A.st2 + (l - 1) * 16,
               A.n2_s + (l - 1) * Dd, A.n2_b + (l - 1) * Dd, m, d);
}

// fill LDS A-fragment buffer (rows 8..15 pre-zeroed once) with h at layer l
__device__ inline void fill_hcur_frag(int l, int t, const Args& A, __bf16* sAbf) {
  for (int i = threadIdx.x; i < 8192; i += NTHR) {
    const int m = i >> 10, d = i & 1023;
    sAbf[afrag_off(m, d)] = (__bf16)hcur_at(l, t, A, m, d);
  }
}

// fill LDS A-fragment buffer with LN(y; stats)
__device__ inline void fill_ln_frag(const Args& A, const float* y, const float* st,
                                    const float* sc, const float* bi, __bf16* sAbf) {
  for (int i = threadIdx.x; i < 8192; i += NTHR) {
    const int m = i >> 10, d = i & 1023;
    sAbf[afrag_off(m, d)] = (__bf16)ln_at(y, st, sc, bi, m, d);
  }
}

// ---------------------------------------------------------------------------
// weight swizzle: fp32 row-major (l,K,N) -> bf16 WMMA-B tiles, (nt,kt) order
// ---------------------------------------------------------------------------
__device__ inline void swizzle(const float* src, __bf16* dst, int K, int N,
                               int nl, long gtid, long TT) {
  const int KT = K >> 5;
  const long per = (long)K * N;
  const long total = per * nl;
  for (long i = gtid; i < total; i += TT) {
    const int l = (int)(i / per);
    const long r = i - (long)l * per;
    const int tile = (int)(r >> 9), inner = (int)(r & 511);
    const int lane = inner >> 4, e = inner & 15;
    const int nt = tile / KT, kt = tile - nt * KT;
    const int k = kt * 32 + (lane >> 4) * 8 + (e >> 3) * 16 + (e & 7);
    const int n = nt * 16 + (lane & 15);
    dst[i] = (__bf16)src[(long)l * per + (long)k * N + n];
  }
}

// ---------------------------------------------------------------------------
// off-by-one-softmax attention, 2 (b,head) tasks per block
// att written directly into A-fragment-swizzled global buffer for phase B
// ---------------------------------------------------------------------------
__device__ inline void attention_phase(int t, const Args& A, float* sred,
                                       float* sscore, float* sQ, float* satt) {
  const int tid = threadIdx.x;
  const int bid = blockIdx.x;
  for (int task = bid * 2; task < bid * 2 + 2; ++task) {
    const int b = task >> 4, head = task & 15;
    if (tid < 64) sQ[tid] = A.qbuf[b * Dd + head * HDd + tid];
    __syncthreads();
    float lmax = 0.f;   // softmax1: max includes 0
    for (int s = tid; s < t; s += NTHR) {
      const float* kk = A.mk + ((size_t)b * Ss + s) * Dd + head * HDd;
      float dot = 0.f;
#pragma unroll
      for (int d = 0; d < HDd; ++d) dot += sQ[d] * kk[d];
      dot *= SCALE;
      sscore[s] = dot;
      lmax = fmaxf(lmax, dot);
    }
    sred[tid] = lmax;
    __syncthreads();
    for (int o = NTHR / 2; o > 0; o >>= 1) {
      if (tid < o) sred[tid] = fmaxf(sred[tid], sred[tid + o]);
      __syncthreads();
    }
    const float smax = sred[0];
    __syncthreads();
    float lsum = 0.f;
    for (int s = tid; s < t; s += NTHR) {
      float e = __expf(sscore[s] - smax);
      sscore[s] = e;
      lsum += e;
    }
    sred[tid] = lsum;
    __syncthreads();
    for (int o = NTHR / 2; o > 0; o >>= 1) {
      if (tid < o) sred[tid] += sred[tid + o];
      __syncthreads();
    }
    const float denom = __expf(-smax) + sred[0];
    __syncthreads();
    {
      const int hd = tid & 63, sq = tid >> 6;
      float acc = 0.f;
      for (int s = sq; s < t; s += 4)
        acc += sscore[s] * A.mv[((size_t)b * Ss + s) * Dd + head * HDd + hd];
      satt[sq * 64 + hd] = acc;
    }
    __syncthreads();
    if (tid < 64) {
      float aa = (satt[tid] + satt[64 + tid] + satt[128 + tid] + satt[192 + tid]) / denom;
      A.attb[afrag_off(b, head * HDd + tid)] = (__bf16)aa;
    }
    __syncthreads();
  }
}

// ---------------------------------------------------------------------------
// persistent kernel: setup, then 512 steps of phased work with grid barriers
// ---------------------------------------------------------------------------
__global__ __launch_bounds__(NTHR)
void FeedbackTransformerModel_60181081751899_kernel(Args A) {
  const int tid = threadIdx.x;
  const int bid = blockIdx.x;
  const long gtid = (long)bid * NTHR + tid;
  const long TT = (long)NBLK * NTHR;

  __shared__ __align__(16) __bf16 sAbf[32 * 512];  // 32KB A-fragment staging (K<=1024)
  __shared__ float sred[1024];                     // 4KB cross-wave reduce
  __shared__ float sscore[512];                    // attention scores
  __shared__ float sQ[64];
  __shared__ float satt[256];

  // zero entire LDS A-fragment buffer once; fills only ever touch rows m<8
  for (int i = tid; i < 32 * 512; i += NTHR) sAbf[i] = (__bf16)0.0f;

  // ---- setup: bf16 swizzled weights + embedded inputs + zero A-frag pads ----
  swizzle(A.q_w,   A.wq,   Dd, Dd, Ll, gtid, TT);
  swizzle(A.out_w, A.wout, Dd, Dd, Ll, gtid, TT);
  swizzle(A.ff1_w, A.wff1, Dd, Ff, Ll, gtid, TT);
  swizzle(A.ff2_w, A.wff2, Ff, Dd, Ll, gtid, TT);
  swizzle(A.k_w,   A.wk,   Dd, Dd, 1,  gtid, TT);
  swizzle(A.v_w,   A.wv,   Dd, Dd, 1,  gtid, TT);
  swizzle(A.lm_w,  A.wlm,  Dd, Vv, 1,  gtid, TT);
  for (long i = gtid; i < (long)Ss * Bb * Dd; i += TT) {
    const int d = (int)(i & 1023);
    const long r = i >> 10;
    const int b = (int)(r & 7), s = (int)(r >> 3);
    A.xseq[i] = A.emb[(size_t)A.ids[b * Ss + s] * Dd + d] + A.pe[(size_t)s * Dd + d];
  }
  for (long i = gtid; i < 32 * 512; i += TT)  A.attb[i] = (__bf16)0.0f;   // zero pads
  for (long i = gtid; i < 128 * 512; i += TT) A.a1b[i]  = (__bf16)0.0f;   // zero pads
  gsync(A.bar);

#pragma unroll 1
  for (int t = 0; t < A.steps; ++t) {
#pragma unroll 1
    for (int l = 0; l < A.layers; ++l) {
      // ---- phase A1: q = h @ q_w + q_b (WMMA, A staged in LDS fragments) ----
      fill_hcur_frag(l, t, A, sAbf);
      if (bid == 0 && tid < 16) A.st1[tid] = 0.f;
      __syncthreads();
      run_tile(sAbf, A.wq + (size_t)l * Dd * Dd, 32, bid, sred, nullptr,
               [&](int m, int col, float acc) -> float {
                 A.qbuf[m * Dd + col] = acc + A.q_b[l * Dd + col];
                 return 0.f;
               });
      gsync(A.bar);

      // ---- phase A2: attention with softmax1 over memory s < t ----
      attention_phase(t, A, sred, sscore, sQ, satt);
      gsync(A.bar);

      // ---- phase B: y1 = h + att @ out_w + out_b ; LN1 stats ----
      // A comes pre-swizzled from attention; residual h recomputed inline.
      run_tile(A.attb, A.wout + (size_t)l * Dd * Dd, 32, bid, sred, A.st1,
               [&](int m, int col, float acc) -> float {
                 float y = acc + A.out_b[l * Dd + col] + hcur_at(l, t, A, m, col);
                 A.y1[m * Dd + col] = y;
                 return y;
               });
      gsync(A.bar);

      // ---- phase C: a1 = relu(hn1 @ ff1_w + ff1_b), written as A-fragments ----
      fill_ln_frag(A, A.y1, A.st1, A.n1_s + l * Dd, A.n1_b + l * Dd, sAbf);
      if (bid == 0 && tid < 16) A.st2[l * 16 + tid] = 0.f;
      __syncthreads();
#pragma unroll 1
      for (int i = 0; i < 4; ++i) {
        run_tile(sAbf, A.wff1 + (size_t)l * Dd * Ff, 32, bid * 4 + i, sred, nullptr,
                 [&](int m, int col, float acc) -> float {
                   float v = fmaxf(acc + A.ff1_b[l * Ff + col], 0.f);
                   A.a1b[afrag_off(m, col)] = (__bf16)v;
                   return 0.f;
                 });
      }
      gsync(A.bar);

      // ---- phase D: y2 = hn1 + a1 @ ff2_w + ff2_b ; LN2 stats ----
      run_tile(A.a1b, A.wff2 + (size_t)l * Ff * Dd, 128, bid, sred, A.st2 + l * 16,
               [&](int m, int col, float acc) -> float {
                 float hn = ln_at(A.y1, A.st1, A.n1_s + l * Dd, A.n1_b + l * Dd, m, col);
                 float y = acc + A.ff2_b[l * Dd + col] + hn;
                 A.y2[(size_t)(l * 8 + m) * Dd + col] = y;
                 return y;
               });
      gsync(A.bar);
    }

    // ---- phase E: LM head + memory write (3 WMMA GEMMs) ----
    if (tid == 0) {   // softmax(mem_w), 5 weights
      float w[5], mx = -1e30f;
#pragma unroll
      for (int i = 0; i < 5; ++i) { w[i] = A.mem_w[i]; mx = fmaxf(mx, w[i]); }
      float s = 0.f;
#pragma unroll
      for (int i = 0; i < 5; ++i) { w[i] = __expf(w[i] - mx); s += w[i]; }
#pragma unroll
      for (int i = 0; i < 5; ++i) sQ[i] = w[i] / s;
    }
    fill_ln_frag(A, A.y2 + (size_t)3 * 8192, A.st2 + 3 * 16,
                 A.n2_s + 3 * Dd, A.n2_b + 3 * Dd, sAbf);   // h4
    __syncthreads();
    run_tile(sAbf, A.wlm, 32, bid, sred, nullptr,
             [&](int m, int col, float acc) -> float {
               A.out[((size_t)m * Ss + t) * Vv + col] = acc + A.lm_b[col];
               return 0.f;
             });
    // memv = softmax(mem_w) . [x_t, h1..h4], staged as A-fragments
    for (int i = tid; i < 8192; i += NTHR) {
      const int m = i >> 10, d = i & 1023;
      float v = sQ[0] * A.xseq[(size_t)t * 8192 + i];
#pragma unroll
      for (int l2 = 0; l2 < 4; ++l2) {
        float mu  = A.st2[(l2 * 8 + m) * 2] * INV_D;
        float var = A.st2[(l2 * 8 + m) * 2 + 1] * INV_D - mu * mu;
        float rs  = rsqrtf(var + 1e-5f);
        v += sQ[1 + l2] * ((A.y2[(size_t)(l2 * 8 + m) * Dd + d] - mu) * rs *
                           A.n2_s[l2 * Dd + d] + A.n2_b[l2 * Dd + d]);
      }
      sAbf[afrag_off(m, d)] = (__bf16)v;
    }
    __syncthreads();
    run_tile(sAbf, A.wk, 32, bid, sred, nullptr,
             [&](int m, int col, float acc) -> float {
               A.mk[((size_t)m * Ss + t) * Dd + col] = acc + A.k_b[col];
               return 0.f;
             });
    run_tile(sAbf, A.wv, 32, bid, sred, nullptr,
             [&](int m, int col, float acc) -> float {
               A.mv[((size_t)m * Ss + t) * Dd + col] = acc + A.v_b[col];
               return 0.f;
             });
    gsync(A.bar);
  }
}

// ---------------------------------------------------------------------------
extern "C" void kernel_launch(void* const* d_in, const int* in_sizes, int n_in,
                              void* d_out, int out_size, void* d_ws, size_t ws_size,
                              hipStream_t stream) {
  (void)in_sizes; (void)n_in; (void)out_size; (void)ws_size;
  Args a;
  a.ids   = (const int*)d_in[0];
  a.emb   = (const float*)d_in[1];
  a.pe    = (const float*)d_in[2];
  a.q_w   = (const float*)d_in[3];
  a.q_b   = (const float*)d_in[4];
  a.out_w = (const float*)d_in[5];
  a.out_b = (const float*)d_in[6];
  a.ff1_w = (const float*)d_in[7];
  a.ff1_b = (const float*)d_in[8];
  a.ff2_w = (const float*)d_in[9];
  a.ff2_b = (const float*)d_in[10];
  a.n1_s  = (const float*)d_in[11];
  a.n1_b  = (const float*)d_in[12];
  a.n2_s  = (const float*)d_in[13];
  a.n2_b  = (const float*)d_in[14];
  a.mem_w = (const float*)d_in[15];
  a.k_w   = (const float*)d_in[16];
  a.k_b   = (const float*)d_in[17];
  a.v_w   = (const float*)d_in[18];
  a.v_b   = (const float*)d_in[19];
  a.lm_w  = (const float*)d_in[20];
  a.lm_b  = (const float*)d_in[21];
  a.out   = (float*)d_out;

  char* p = (char*)d_ws;
  size_t off = 0;
  auto take = [&](size_t bytes) -> char* {
    char* r = p + off;
    off += (bytes + 255) & ~(size_t)255;
    return r;
  };
  a.bar  = (unsigned*)take(4096);
  a.wq   = (__bf16*)take((size_t)Ll * Dd * Dd * 2);
  a.wout = (__bf16*)take((size_t)Ll * Dd * Dd * 2);
  a.wff1 = (__bf16*)take((size_t)Ll * Dd * Ff * 2);
  a.wff2 = (__bf16*)take((size_t)Ll * Ff * Dd * 2);
  a.wk   = (__bf16*)take((size_t)Dd * Dd * 2);
  a.wv   = (__bf16*)take((size_t)Dd * Dd * 2);
  a.wlm  = (__bf16*)take((size_t)Dd * Vv * 2);
  a.xseq = (float*)take((size_t)Ss * Bb * Dd * 4);
  a.mk   = (float*)take((size_t)Bb * Ss * Dd * 4);
  a.mv   = (float*)take((size_t)Bb * Ss * Dd * 4);
  a.qbuf = (float*)take((size_t)Bb * Dd * 4);
  a.attb = (__bf16*)take((size_t)32 * 512 * 2);    // att, A-fragment swizzled (16 rows)
  a.y1   = (float*)take((size_t)Bb * Dd * 4);
  a.st1  = (float*)take(256);
  a.a1b  = (__bf16*)take((size_t)128 * 512 * 2);   // ffn hidden, A-fragment swizzled
  a.y2   = (float*)take((size_t)Ll * Bb * Dd * 4);
  a.st2  = (float*)take(256);
  a.steps = Ss;
  a.layers = Ll;

  hipMemsetAsync(d_ws, 0, 4096, stream);   // barrier state, deterministic per call
  hipLaunchKernelGGL(FeedbackTransformerModel_60181081751899_kernel,
                     dim3(NBLK), dim3(NTHR), 0, stream, a);
}
